// PSNv1_76716705841951
// MI455X (gfx1250) — matrix operand order repaired and checked by reading, not compiled, not run on hardware
//
#include <hip/hip_runtime.h>

#define B_  16
#define M_  8192
#define D_  64
#define S_  512
#define NL  32
#define C0_ 32
#define C1_ 128
#define EPS_ 1e-5f

typedef __attribute__((ext_vector_type(16))) __bf16 v16bf;
typedef __attribute__((ext_vector_type(8)))  __bf16 v8bf;
typedef __attribute__((ext_vector_type(8)))  float  v8f;

static __device__ __forceinline__ unsigned short f32_to_bf16(float f) {
  unsigned int u = __float_as_uint(f);
  u += 0x7FFFu + ((u >> 16) & 1u);       // round to nearest even
  return (unsigned short)(u >> 16);
}

// ---------------------------------------------------------------------------
// Phase 1: geometry + MLP 5->32->128 (BN folded), output bf16 features
// ---------------------------------------------------------------------------
__global__ void __launch_bounds__(256)
feat_kernel(const float* __restrict__ coord,
            const float* __restrict__ W0, const float* __restrict__ b0,
            const float* __restrict__ g0, const float* __restrict__ be0,
            const float* __restrict__ rm0, const float* __restrict__ rv0,
            const float* __restrict__ W1, const float* __restrict__ b1,
            const float* __restrict__ g1, const float* __restrict__ be1,
            const float* __restrict__ rm1, const float* __restrict__ rv1,
            unsigned short* __restrict__ X2) {
  __shared__ float sW0[C0_ * 5];
  __shared__ float sS0[C0_], sB0[C0_];
  __shared__ float sW1[C1_ * C0_];
  __shared__ float sS1[C1_], sB1[C1_];
  const int tid = threadIdx.x;

  for (int i = tid; i < C0_ * 5; i += 256) sW0[i] = W0[i];
  if (tid < C0_) {
    float sc = g0[tid] * rsqrtf(rv0[tid] + EPS_);
    sS0[tid] = sc;
    sB0[tid] = (b0[tid] - rm0[tid]) * sc + be0[tid];
  }
  for (int i = tid; i < C1_ * C0_; i += 256) sW1[i] = W1[i];
  if (tid < C1_) {
    float sc = g1[tid] * rsqrtf(rv1[tid] + EPS_);
    sS1[tid] = sc;
    sB1[tid] = (b1[tid] - rm1[tid]) * sc + be1[tid];
  }
  __syncthreads();

  const int idx = blockIdx.x * 256 + tid;          // 0 .. B*M-1
  float x = coord[idx * 3 + 0];
  float y = coord[idx * 3 + 1];
  float z = coord[idx * 3 + 2];
  float r  = sqrtf(x * x + y * y + z * z);
  float th = acosf(z / r);
  float fi = atan2f(y, x);
  float in5[5] = {x, y, z, th, fi};

  float h0[C0_];
  #pragma unroll
  for (int o = 0; o < C0_; ++o) {
    float acc = 0.f;
    #pragma unroll
    for (int c = 0; c < 5; ++c) acc += sW0[o * 5 + c] * in5[c];
    h0[o] = fmaxf(acc * sS0[o] + sB0[o], 0.f);
  }

  unsigned short* outp = X2 + (size_t)idx * C1_;
  for (int o = 0; o < C1_; ++o) {
    float acc = 0.f;
    #pragma unroll
    for (int c = 0; c < C0_; ++c) acc += sW1[o * C0_ + c] * h0[c];
    outp[o] = f32_to_bf16(fmaxf(acc * sS1[o] + sB1[o], 0.f));
  }
}

// ---------------------------------------------------------------------------
// W2 f32 -> bf16
// ---------------------------------------------------------------------------
__global__ void cvt_kernel(const float* __restrict__ W2,
                           unsigned short* __restrict__ W2b) {
  int i = blockIdx.x * 256 + threadIdx.x;
  W2b[i] = f32_to_bf16(W2[i]);
}

// ---------------------------------------------------------------------------
// Phase 2: Q[b,s,m] = sigmoid(X2[b,m,:] . W2[s,:] + b2[s]) via WMMA bf16
//   WMMA M-dim = s (A = W2 tile), N-dim = m (B = X2^T tile), K = 128 (4x32)
//   grid: (M/128, S/128, B), block 256 = 8 waves; wave owns one 16-s stripe,
//   keeps its 4 A fragments resident, sweeps 8 m-tiles.
// ---------------------------------------------------------------------------
__global__ void __launch_bounds__(256)
gemm_sigmoid_kernel(const unsigned short* __restrict__ X2,
                    const unsigned short* __restrict__ W2b,
                    const float* __restrict__ b2,
                    float* __restrict__ Q) {
  const int lane  = threadIdx.x & 31;
  const int wave  = threadIdx.x >> 5;
  const int b     = blockIdx.z;
  const int s0    = blockIdx.y * 128 + wave * 16;
  const int mbase = blockIdx.x * 128;
  const int row   = lane & 15;   // A: s-row within tile; B/D: m-column
  const int half  = lane >> 4;   // K-half select / D row-half select

  // A fragments: W2 rows s0..s0+15, all 4 K-steps (resident across m-tiles)
  v16bf afrag[4];
  const unsigned short* wrow = W2b + (size_t)(s0 + row) * C1_ + half * 16;
  #pragma unroll
  for (int kk = 0; kk < 4; ++kk) {
    afrag[kk] = *reinterpret_cast<const v16bf*>(wrow + kk * 32);
  }

  float bias[8];
  #pragma unroll
  for (int rI = 0; rI < 8; ++rI) bias[rI] = b2[s0 + rI + half * 8];

  for (int mt = 0; mt < 8; ++mt) {
    const int m0 = mbase + mt * 16;
    const unsigned short* xrow =
        X2 + ((size_t)b * M_ + (m0 + row)) * C1_ + half * 16;
    v8f acc = {};
    #pragma unroll
    for (int kk = 0; kk < 4; ++kk) {
      v16bf bfrag = *reinterpret_cast<const v16bf*>(xrow + kk * 32);
      acc = __builtin_amdgcn_wmma_f32_16x16x32_bf16(
          false, afrag[kk], false, bfrag, (short)0, acc, false, false);
    }
    // D layout: elem rI, lane -> s = s0 + rI + 8*half, m = m0 + row
    #pragma unroll
    for (int rI = 0; rI < 8; ++rI) {
      const int srow = s0 + rI + half * 8;
      float v = acc[rI] + bias[rI];
      // sigmoid with HW approx reciprocal (v_exp_f32 + v_rcp_f32)
      Q[((size_t)b * S_ + srow) * (size_t)M_ + (m0 + row)] =
          __builtin_amdgcn_rcpf(1.0f + __expf(-v));
    }
  }
}

// ---------------------------------------------------------------------------
// Phase 3: per (b,s) row: top-32 of Q row, then gather points/features.
// Incremental argmax: each thread keeps (best,idx) over its 32 strided
// elements in registers; per round: wave32 shuffle argmax (no barrier),
// 8-way combine by thread 0, only the owner thread rescans its subset.
// Tie-break everywhere: lower index wins (matches jax.lax.top_k).
// ---------------------------------------------------------------------------
__global__ void __launch_bounds__(256)
topk_gather_kernel(const float* __restrict__ Q,
                   const float* __restrict__ coord,
                   const float* __restrict__ feat,
                   float* __restrict__ sp, float* __restrict__ gp,
                   float* __restrict__ sf, float* __restrict__ gf) {
  __shared__ float vals[M_];
  __shared__ float wv[8];
  __shared__ int   wi[8];
  __shared__ int   chosen[NL];
  __shared__ int   winIdx;
  const int tid  = threadIdx.x;
  const int lane = tid & 31;
  const int wv_id = tid >> 5;
  const int rowI = blockIdx.x;                 // b*S + s
  const int b    = rowI / S_;
  const float* q = Q + (size_t)rowI * M_;

  for (int i = tid; i < M_; i += 256) vals[i] = q[i];
  __syncthreads();

  // initial per-thread best over strided subset (i === tid mod 256)
  float best = -3.0e38f;
  int   bi   = M_;
  for (int i = tid; i < M_; i += 256) {
    float v = vals[i];
    if (v > best) { best = v; bi = i; }
  }

  for (int sel = 0; sel < NL; ++sel) {
    // wave32 argmax reduction via shuffles
    float rv = best;
    int   ri = bi;
    #pragma unroll
    for (int off = 16; off > 0; off >>= 1) {
      float ov = __shfl_xor(rv, off, 32);
      int   oi = __shfl_xor(ri, off, 32);
      if (ov > rv || (ov == rv && oi < ri)) { rv = ov; ri = oi; }
    }
    if (lane == 0) { wv[wv_id] = rv; wi[wv_id] = ri; }
    __syncthreads();
    if (tid == 0) {
      float fv = wv[0]; int fi = wi[0];
      #pragma unroll
      for (int w = 1; w < 8; ++w) {
        float ov = wv[w]; int oi = wi[w];
        if (ov > fv || (ov == fv && oi < fi)) { fv = ov; fi = oi; }
      }
      chosen[sel] = fi;
      vals[fi]    = -3.0e38f;
      winIdx      = fi;
    }
    __syncthreads();
    const int win = winIdx;
    if ((win & 255) == tid) {          // owner rescans its 32 elements
      best = -3.0e38f; bi = M_;
      for (int i = tid; i < M_; i += 256) {
        float v = vals[i];
        if (v > best) { best = v; bi = i; }
      }
    }
  }

  // gathers
  for (int t = tid; t < NL * D_; t += 256) {
    int k = t >> 6, c = t & 63;
    gf[((size_t)rowI * NL + k) * D_ + c] =
        feat[((size_t)b * M_ + chosen[k]) * D_ + c];
  }
  for (int t = tid; t < NL * 3; t += 256) {
    int k = t / 3, c = t % 3;
    gp[((size_t)rowI * NL + k) * 3 + c] =
        coord[((size_t)b * M_ + chosen[k]) * 3 + c];
  }
  if (tid < D_) sf[(size_t)rowI * D_ + tid] =
      feat[((size_t)b * M_ + chosen[0]) * D_ + tid];
  if (tid < 3)  sp[(size_t)rowI * 3 + tid] =
      coord[((size_t)b * M_ + chosen[0]) * 3 + tid];
}

// ---------------------------------------------------------------------------
extern "C" void kernel_launch(void* const* d_in, const int* in_sizes, int n_in,
                              void* d_out, int out_size, void* d_ws, size_t ws_size,
                              hipStream_t stream) {
  (void)in_sizes; (void)n_in; (void)out_size; (void)ws_size;
  const float* coord = (const float*)d_in[0];
  const float* feat  = (const float*)d_in[1];
  const float* W0  = (const float*)d_in[2];
  const float* b0  = (const float*)d_in[3];
  const float* g0  = (const float*)d_in[4];
  const float* be0 = (const float*)d_in[5];
  const float* rm0 = (const float*)d_in[6];
  const float* rv0 = (const float*)d_in[7];
  const float* W1  = (const float*)d_in[8];
  const float* b1  = (const float*)d_in[9];
  const float* g1  = (const float*)d_in[10];
  const float* be1 = (const float*)d_in[11];
  const float* rm1 = (const float*)d_in[12];
  const float* rv1 = (const float*)d_in[13];
  const float* W2  = (const float*)d_in[14];
  const float* b2  = (const float*)d_in[15];

  float* out = (float*)d_out;
  float* sp = out;                                        // (16,512,3)
  float* gp = out + 24576;                                // (16,512,32,3)
  float* sf = out + 24576 + 786432;                       // (16,512,64)
  float* gf = out + 24576 + 786432 + 524288;              // (16,512,32,64)
  float* Q  = out + 24576 + 786432 + 524288 + 16777216;   // (16,512,8192)

  unsigned short* X2b = (unsigned short*)d_ws;            // 32 MB bf16 features
  unsigned short* W2b = X2b + (size_t)B_ * M_ * C1_;      // 128 KB bf16 W2

  feat_kernel<<<dim3((B_ * M_) / 256), 256, 0, stream>>>(
      coord, W0, b0, g0, be0, rm0, rv0, W1, b1, g1, be1, rm1, rv1, X2b);
  cvt_kernel<<<dim3((S_ * C1_) / 256), 256, 0, stream>>>(W2, W2b);
  gemm_sigmoid_kernel<<<dim3(M_ / 128, S_ / 128, B_), 256, 0, stream>>>(
      X2b, W2b, b2, Q);
  topk_gather_kernel<<<dim3(B_ * S_), 256, 0, stream>>>(
      Q, coord, feat, sp, gp, sf, gf);
}
